// BinarizeConv2d_29523605193204
// MI455X (gfx1250) — 8-bit matrix operand order repaired and checked by reading, compiled, not run on hardware
//
#include <hip/hip_runtime.h>


typedef int v8i __attribute__((ext_vector_type(8)));
typedef int v4i_g __attribute__((vector_size(16)));

#define C_IN   256
#define C_OUT  256
#define HH     56
#define WW     56
#define BATCH  32
#define HP     58
#define WPAD   58
#define PIX    (HH*WW)
#define KDIM   (C_IN*9)
#define LDS_BUF 32768

#define BA_BYTES     ((size_t)BATCH*HP*WPAD*C_IN)
#define WPACK_BYTES  ((size_t)9*4*16*32*32)
#define WSIGN_BYTES  ((size_t)C_OUT*C_IN*9)
#define OFF_BA   ((size_t)0)
#define OFF_WP   (OFF_BA + BA_BYTES)
#define OFF_WS   (OFF_WP + WPACK_BYTES)
#define OFF_F    (OFF_WS + WSIGN_BYTES)

__device__ __forceinline__ void async_copy_b128(const unsigned char* g, unsigned char* l) {
#if __has_builtin(__builtin_amdgcn_global_load_async_to_lds_b128)
    __builtin_amdgcn_global_load_async_to_lds_b128(
        (__attribute__((address_space(1))) v4i_g*)g,
        (__attribute__((address_space(3))) v4i_g*)l, 0, 0);
#else
    unsigned lofs = (unsigned)(unsigned long long)
        (__attribute__((address_space(3))) unsigned char*)l;
    unsigned long long ga = (unsigned long long)g;
    asm volatile("global_load_async_to_lds_b128 %0, %1, off"
                 :: "v"(lofs), "v"(ga) : "memory");
#endif
}

__device__ __forceinline__ void wait_async0() {
#if __has_builtin(__builtin_amdgcn_s_wait_asynccnt)
    __builtin_amdgcn_s_wait_asynccnt(0);
#else
    asm volatile("s_wait_asynccnt 0x0" ::: "memory");
#endif
}

__global__ void k_zero(uint4* __restrict__ p, int n16) {
    int i = blockIdx.x * blockDim.x + threadIdx.x;
    if (i < n16) p[i] = make_uint4(0u, 0u, 0u, 0u);
}

__global__ void k_wprep(const float* __restrict__ W, const float* __restrict__ RV,
                        const float* __restrict__ alpha,
                        signed char* __restrict__ wsign, float* __restrict__ f) {
    const int co = blockIdx.x;
    const int ci = threadIdx.x;
    const float r0 = RV[0], r1 = RV[1], r2 = RV[2], r3 = RV[3];
    const size_t KSTR = (size_t)C_OUT * C_IN * 9;
    const size_t base = ((size_t)co * C_IN + ci) * 9;
    float asum = 0.f;
#pragma unroll
    for (int s = 0; s < 9; ++s) {
        float rw = r0 * W[base + s] + r1 * W[KSTR + base + s] +
                   r2 * W[2 * KSTR + base + s] + r3 * W[3 * KSTR + base + s];
        asum += fabsf(rw);
        wsign[base + s] = (rw > 0.f) ? (signed char)1
                        : ((rw < 0.f) ? (signed char)-1 : (signed char)0);
    }
    __shared__ float red[256];
    red[ci] = asum;
    __syncthreads();
    for (int st = 128; st > 0; st >>= 1) {
        if (ci < st) red[ci] += red[ci + st];
        __syncthreads();
    }
    if (ci == 0) f[co] = (red[0] / (float)KDIM) * alpha[co];
}

__global__ void k_wpack(const signed char* __restrict__ wsign, int* __restrict__ wpack) {
    int t = blockIdx.x * blockDim.x + threadIdx.x;
    if (t >= 9 * 4 * 16 * 32) return;
    const int lane    = t & 31;
    const int co_tile = (t >> 5) & 15;
    const int chunk   = (t >> 9) & 3;
    const int s       = t >> 11;
    const int co  = co_tile * 16 + (lane & 15);
    const int khi = (lane < 16) ? 0 : 16;
#pragma unroll
    for (int v = 0; v < 8; ++v) {
        unsigned int d = 0;
#pragma unroll
        for (int j = 0; j < 4; ++j) {
            int K  = (v >> 2) * 32 + (v & 3) * 4 + khi + j;
            int ci = chunk * 64 + K;
            unsigned char bb = (unsigned char)wsign[((size_t)co * C_IN + ci) * 9 + s];
            d |= ((unsigned int)bb) << (8 * j);
        }
        wpack[(size_t)t * 8 + v] = (int)d;
    }
}

__global__ void k_binact(const float* __restrict__ x, unsigned char* __restrict__ ba) {
    const int wt = blockIdx.x, h = blockIdx.y, b = blockIdx.z;
    const int t = threadIdx.x;
    __shared__ signed char tile[64 * 36];
    const int w0 = wt * 32;

    for (int cb = 0; cb < C_IN; cb += 64) {
        const int w_l = t & 31;
        const int ci0 = t >> 5;
        const int w = w0 + w_l;
#pragma unroll
        for (int it = 0; it < 8; ++it) {
            const int ci_l = it * 8 + ci0;
            signed char sg = 0;
            if (w < WW) {
                float xv = x[(((size_t)b * C_IN + (cb + ci_l)) * HH + h) * WW + w];
                sg = (xv > 0.f) ? (signed char)1
                   : ((xv < 0.f) ? (signed char)-1 : (signed char)0);
            }
            tile[ci_l * 36 + w_l] = sg;
        }
        __syncthreads();
        const int w_l2 = t >> 3;
        const int part = t & 7;
        const int w2 = w0 + w_l2;
        if (w2 < WW) {
            unsigned long long v = 0ull;
#pragma unroll
            for (int j = 0; j < 8; ++j)
                v |= ((unsigned long long)(unsigned char)tile[(part * 8 + j) * 36 + w_l2])
                     << (8 * j);
            *(unsigned long long*)(ba + (((size_t)b * HP + (h + 1)) * WPAD + (w2 + 1)) * C_IN
                                      + cb + part * 8) = v;
        }
        __syncthreads();
    }
}

__global__ void __launch_bounds__(256)
k_conv(const unsigned char* __restrict__ ba, const int* __restrict__ wpack,
       const float* __restrict__ f, float* __restrict__ out) {
    __shared__ unsigned char smem[2 * LDS_BUF];
    const int lane = threadIdx.x & 31;
    const int wave = threadIdx.x >> 5;
    const int b    = blockIdx.z;
    const int co_tile = blockIdx.y * 8 + wave;
    const int p0 = blockIdx.x * 128;

    v8i acc[8];
#pragma unroll
    for (int i = 0; i < 8; ++i)
#pragma unroll
        for (int j = 0; j < 8; ++j) acc[i][j] = 0;

    int pf = p0 + (threadIdx.x >> 1);
    if (pf > PIX - 1) pf = PIX - 1;
    const int fh = pf / WW, fw = pf % WW;
    const int half = (threadIdx.x & 1) * 128;
    unsigned char* ldst_base = smem + (size_t)(threadIdx.x >> 1) * 256 + half;

    auto issue_shift = [&](int s, int buf) {
        const int kh = s / 3, kw = s % 3;
        const unsigned char* src = ba +
            (((size_t)b * HP + (fh + kh)) * WPAD + (fw + kw)) * C_IN + half;
        unsigned char* dst = ldst_base + buf * LDS_BUF;
#pragma unroll
        for (int i = 0; i < 8; ++i)
            async_copy_b128(src + i * 16, dst + i * 16);
    };

    issue_shift(0, 0);

    for (int s = 0; s < 9; ++s) {
        wait_async0();
        __syncthreads();
        if (s < 8) issue_shift(s + 1, (s + 1) & 1);

        v8i afragS[4];
#pragma unroll
        for (int c = 0; c < 4; ++c)
            afragS[c] = *(const v8i*)(wpack +
                (size_t)((((s * 4 + c) * 16 + co_tile) * 32 + lane) * 8));

        const unsigned char* sbase = smem + (s & 1) * LDS_BUF +
            (size_t)(lane & 15) * 256 + ((lane < 16) ? 0 : 16);
#pragma unroll
        for (int chunk = 0; chunk < 4; ++chunk) {
            const unsigned char* cbp = sbase + chunk * 64;
            union { int4 q[2]; v8i v; } bb[8];
#pragma unroll
            for (int pt = 0; pt < 8; ++pt) {
                bb[pt].q[0] = *(const int4*)(cbp + pt * 4096);
                bb[pt].q[1] = *(const int4*)(cbp + pt * 4096 + 32);
            }
#pragma unroll
            for (int pt = 0; pt < 8; ++pt)
                acc[pt] = __builtin_amdgcn_wmma_i32_16x16x64_iu8(
                     true, afragS[chunk],  true, bb[pt].v,
                    acc[pt],  false,  false);
        }
    }

    const int coB = co_tile * 16 + ((lane >> 4) << 3);
    float fv[8];
#pragma unroll
    for (int r = 0; r < 8; ++r) fv[r] = f[coB + r];
    const int pl = p0 + (lane & 15);
#pragma unroll
    for (int pt = 0; pt < 8; ++pt) {
        const int p = pl + pt * 16;
        if (p < PIX) {
#pragma unroll
            for (int r = 0; r < 8; ++r)
                out[((size_t)b * C_OUT + coB + r) * PIX + p] = fv[r] * (float)acc[pt][r];
        }
    }
}

extern "C" void kernel_launch(void* const* d_in, const int* in_sizes, int n_in,
                              void* d_out, int out_size, void* d_ws, size_t ws_size,
                              hipStream_t stream) {
    const float* x     = (const float*)d_in[0];
    const float* W     = (const float*)d_in[1];
    const float* RV    = (const float*)d_in[2];
    const float* alpha = (const float*)d_in[3];
    float* out = (float*)d_out;

    unsigned char* ws    = (unsigned char*)d_ws;
    unsigned char* ba    = ws + OFF_BA;
    int*           wpack = (int*)(ws + OFF_WP);
    signed char*   wsign = (signed char*)(ws + OFF_WS);
    float*         f     = (float*)(ws + OFF_F);

    (void)in_sizes; (void)n_in; (void)out_size; (void)ws_size;

    const int n16 = (int)(BA_BYTES / 16);
    k_zero  <<<(n16 + 255) / 256, 256, 0, stream>>>((uint4*)ba, n16);
    k_wprep <<<C_OUT, 256, 0, stream>>>(W, RV, alpha, wsign, f);
    k_wpack <<<(9 * 4 * 16 * 32 + 255) / 256, 256, 0, stream>>>(wsign, wpack);
    k_binact<<<dim3(2, HH, BATCH), 256, 0, stream>>>(x, ba);
    k_conv  <<<dim3(25, 2, BATCH), 256, 0, stream>>>(ba, wpack, f, out);
}
